// BergmanMatrixLayer_24584392802727
// MI455X (gfx1250) — matrix operand order repaired and checked by reading, not compiled, not run on hardware
//
#include <hip/hip_runtime.h>
#include <hip/hip_bf16.h>

#define B_    8
#define T_    2048
#define HID_  1024
#define NH_   16
#define MD_   16
#define NMAT_ 4096   // NH*MD*MD
#define XF_   512    // NH*MD*2

typedef __attribute__((ext_vector_type(16))) __bf16 v16bf;
typedef __attribute__((ext_vector_type(8)))  __bf16 v8bf;
typedef __attribute__((ext_vector_type(4)))  __bf16 v4bf;
typedef __attribute__((ext_vector_type(8)))  float  v8f;
typedef __attribute__((ext_vector_type(4)))  float  v4f;
typedef __attribute__((ext_vector_type(4)))  int    v4i;

// ---- CDNA5 async global->LDS path (guarded so compile can never break) ----
#if defined(__gfx1250__) && __has_builtin(__builtin_amdgcn_global_load_async_to_lds_b128)
#define HAVE_ASYNC_LDS 1
#else
#define HAVE_ASYNC_LDS 0
#endif

typedef __attribute__((address_space(1))) v4i* glb_v4i_p;   // non-const: matches builtin
typedef __attribute__((address_space(3))) v4i* lds_v4i_p;

__device__ __forceinline__ void async_copy16(const void* gsrc, void* ldst) {
#if HAVE_ASYNC_LDS
    // generic->AS1 / generic->AS3 via integer cast;
    // low 32 bits of a generic LDS pointer are the LDS offset (aperture rule)
    __builtin_amdgcn_global_load_async_to_lds_b128(
        (glb_v4i_p)(__UINTPTR_TYPE__)gsrc,
        (lds_v4i_p)(__UINTPTR_TYPE__)ldst,
        0, 0);
#endif
}

__device__ __forceinline__ void wait_async_then_barrier() {
#if HAVE_ASYNC_LDS
#if __has_builtin(__builtin_amdgcn_s_wait_asynccnt)
    __builtin_amdgcn_s_wait_asynccnt(0);
#else
    asm volatile("s_wait_asynccnt 0x0" ::: "memory");
#endif
#endif
    __syncthreads();
}

// Transposed B staging: each thread owns a 4(k) x 8(n) sub-block.
// Packs 4 k's per n into one 8-byte ds_store_b64 (vs 32x ds_store_b16).
__device__ __forceinline__ void stageB_t(__bf16 (*lBt)[32], const float* Bsrc,
                                         int ldb, int tid)
{
    const int kq = (tid & 7) * 4;     // 4 consecutive k
    const int ns = (tid >> 3) * 8;    // 8 consecutive n
    const float* s0 = Bsrc + ns;
    v4f r[4][2];
    #pragma unroll
    for (int kr = 0; kr < 4; ++kr) {
        r[kr][0] = *(const v4f*)(s0 + (size_t)(kq + kr) * ldb);
        r[kr][1] = *(const v4f*)(s0 + (size_t)(kq + kr) * ldb + 4);
    }
    #pragma unroll
    for (int i = 0; i < 8; ++i) {
        v4bf p;
        #pragma unroll
        for (int kr = 0; kr < 4; ++kr)
            p[kr] = (__bf16)r[kr][i >> 2][i & 3];
        *(v4bf*)&lBt[ns + i][kq] = p;   // ds_store_b64, 8B aligned
    }
}

// Shared WMMA micro-kernel: one wave, A frag from lA row mt*16.., 8 B tiles.
__device__ __forceinline__ void wmma_strip(const __bf16 (*lA)[32],
                                           const __bf16 (*lBt)[32],
                                           int mt, int nh2, int half, int l16,
                                           v8f acc[8])
{
    v16bf afrag;
    const int mr = mt * 16 + l16;
    #pragma unroll
    for (int p = 0; p < 8; ++p) {   // contiguous K order: k = 16*half + s (matches B)
        const int kb = half * 16 + p * 2;
        afrag[2 * p]     = lA[mr][kb];
        afrag[2 * p + 1] = lA[mr][kb + 1];
    }
    v16bf bfr[8];                   // preload ALL B frags -> partial dscnt waits
    #pragma unroll
    for (int j = 0; j < 8; ++j)
        bfr[j] = *(const v16bf*)&lBt[(nh2 * 8 + j) * 16 + l16][half * 16];
    #pragma unroll
    for (int j = 0; j < 8; ++j)     // back-to-back WMMAs on the XDL pipe
        acc[j] = __builtin_amdgcn_wmma_f32_16x16x32_bf16(
                     false, afrag, false, bfr[j], (short)0, acc[j], false, false);
}

// ---------------------------------------------------------------------------
// Kernel 1: m = hs @ W_mat + b_mat, per-head Frobenius normalization.
// 64x256 tile / WG, double-buffered LDS, one barrier per K step.
// ---------------------------------------------------------------------------
__global__ __launch_bounds__(256) void k_gemm1_fro(const float* __restrict__ hs,
                                                   const float* __restrict__ Wm,
                                                   const float* __restrict__ bm,
                                                   __bf16* __restrict__ mnorm)
{
    __shared__ __bf16 lA[2][64][32];
    __shared__ __bf16 lBt[2][256][32];
    __shared__ float  lC[64][257];
    __shared__ float  rowss[64];

    const int m0   = blockIdx.x * 64;
    const int n0   = blockIdx.y * 256;
    const int tid  = threadIdx.x;
    const int lane = tid & 31;
    const int wave = tid >> 5;
    const int mt   = wave & 3;
    const int nh2  = wave >> 2;
    const int half = lane >> 4;
    const int l16  = lane & 15;

    const int ra = tid >> 2, ca = (tid & 3) * 8;    // A staging coords

    auto stage = [&](int buf, int k0) {
        const float* srcA = hs + (size_t)(m0 + ra) * HID_ + k0 + ca;
        #pragma unroll
        for (int i = 0; i < 8; ++i) lA[buf][ra][ca + i] = (__bf16)srcA[i];
        stageB_t(lBt[buf], Wm + (size_t)k0 * NMAT_ + n0, NMAT_, tid);
        if (k0 + 32 < HID_) {
            __builtin_prefetch(srcA + 32, 0, 1);
            __builtin_prefetch(Wm + (size_t)(k0 + 32) * NMAT_ + n0 + (tid >> 3) * 8, 0, 1);
        }
    };

    v8f acc[8] = {};
    stage(0, 0);
    const int NK = HID_ / 32;
    for (int kk = 0; kk < NK; ++kk) {
        const int cur = kk & 1;
        __syncthreads();                       // buf[cur] ready; buf[cur^1] drained
        if (kk + 1 < NK) stage(cur ^ 1, (kk + 1) * 32);
        wmma_strip(lA[cur], lBt[cur], mt, nh2, half, l16, acc);
    }

    // bias + park fp32 results in LDS (C/D layout)
    #pragma unroll
    for (int j = 0; j < 8; ++j) {
        const int n    = (nh2 * 8 + j) * 16 + l16;
        const float bv = bm[n0 + n];
        #pragma unroll
        for (int r = 0; r < 8; ++r)
            lC[mt * 16 + (half ? r + 8 : r)][n] = acc[j][r] + bv;
    }
    __syncthreads();

    {   // Frobenius sum of squares per row (4 lanes per row)
        const int r = tid >> 2, q = tid & 3;
        float s = 0.f;
        #pragma unroll
        for (int c = 0; c < 64; ++c) { float v = lC[r][q * 64 + c]; s += v * v; }
        s += __shfl_xor(s, 1, 32);
        s += __shfl_xor(s, 2, 32);
        if (q == 0) rowss[r] = s;
    }
    __syncthreads();
    {   // scale by sqrt(MD)/(fro+eps), emit bf16
        const int r = tid >> 2, q = tid & 3;
        const float scale = 4.0f / (sqrtf(rowss[r]) + 1e-5f);
        __bf16* dst = mnorm + (size_t)(m0 + r) * NMAT_ + n0 + q * 64;
        #pragma unroll
        for (int c = 0; c < 64; ++c) dst[c] = (__bf16)(lC[r][q * 64 + c] * scale);
    }
}

// ---------------------------------------------------------------------------
// Kernel 2: sequential normalized matvec scan; one wave per (sequence, dir),
// 2-deep software prefetch so the serial chain is compute-bound, not L2-bound.
// ---------------------------------------------------------------------------
struct Row16 { v8bf lo, hi; };

__global__ __launch_bounds__(256) void k_scan(const __bf16* __restrict__ mnorm,
                                              __bf16* __restrict__ x)
{
    const int w    = blockIdx.x * 8 + (threadIdx.x >> 5);  // 0..255
    const int dir  = w & 1;
    const int seq  = w >> 1;          // b*NH + nh
    const int b    = seq >> 4;
    const int nh   = seq & 15;
    const int lane = threadIdx.x & 31;
    const int i    = lane & 15;       // matrix row owned by this lane

    float v[16];
    #pragma unroll
    for (int j = 0; j < 16; ++j) v[j] = 0.f;
    v[0] = 1.f;                       // init_type == 'one'

    const __bf16* base = mnorm + (size_t)nh * 256 + (size_t)i * 16;
    __bf16* xb = x + (size_t)nh * 32 + dir * 16 + i;

    auto ld = [&](int s) {
        const int t = dir ? (T_ - 1 - s) : s;
        const __bf16* Mrow = base + (size_t)(b * T_ + t) * NMAT_;
        Row16 r;
        r.lo = *(const v8bf*)(Mrow);       // global_load_b128
        r.hi = *(const v8bf*)(Mrow + 8);   // global_load_b128
        return r;
    };

    Row16 pa = ld(0);
    Row16 pb = ld(1);
    #pragma unroll 2
    for (int s = 0; s < T_; ++s) {
        const Row16 c = pa;
        pa = pb;
        if (s + 2 < T_) pb = ld(s + 2);   // address independent of v -> pipelines
        float acc = 0.f;
        #pragma unroll
        for (int j = 0; j < 8; ++j) acc += (float)c.lo[j] * v[j];
        #pragma unroll
        for (int j = 0; j < 8; ++j) acc += (float)c.hi[j] * v[8 + j];
        float ss = acc * acc;             // L2 norm over the 16 rows
        ss += __shfl_xor(ss, 1, 16);
        ss += __shfl_xor(ss, 2, 16);
        ss += __shfl_xor(ss, 4, 16);
        ss += __shfl_xor(ss, 8, 16);
        const float nv = acc / (sqrtf(ss) + 1e-6f);
        #pragma unroll
        for (int j = 0; j < 16; ++j) v[j] = __shfl(nv, j, 16);
        const int t = dir ? (T_ - 1 - s) : s;
        if (lane < 16)
            xb[(size_t)(b * T_ + t) * XF_] = (__bf16)nv;  // (B,T,NH,2*MD) layout
    }
}

// ---------------------------------------------------------------------------
// Kernel 3: out = gelu(x @ W_out + b_out), exact erf form.
// A tile (already bf16) staged via async global->LDS (ASYNCcnt DMA path).
// ---------------------------------------------------------------------------
__global__ __launch_bounds__(256) void k_gemm2_gelu(const __bf16* __restrict__ x,
                                                    const float* __restrict__ Wo,
                                                    const float* __restrict__ bo,
                                                    float* __restrict__ out)
{
    __shared__ __bf16 lA[2][64][32];
    __shared__ __bf16 lBt[2][256][32];

    const int m0   = blockIdx.x * 64;
    const int n0   = blockIdx.y * 256;
    const int tid  = threadIdx.x;
    const int lane = tid & 31;
    const int wave = tid >> 5;
    const int mt   = wave & 3;
    const int nh2  = wave >> 2;
    const int half = lane >> 4;
    const int l16  = lane & 15;

    const int ra = tid >> 2, ca = (tid & 3) * 8;

    auto stage = [&](int buf, int k0) {
        const __bf16* srcA = x + (size_t)(m0 + ra) * XF_ + k0 + ca;
#if HAVE_ASYNC_LDS
        // 16B per thread, ASYNCcnt-tracked DMA into LDS (no VGPR round trip)
        async_copy16(srcA, &lA[buf][ra][ca]);
#else
        #pragma unroll
        for (int i = 0; i < 8; ++i) lA[buf][ra][ca + i] = srcA[i];
#endif
        stageB_t(lBt[buf], Wo + (size_t)k0 * HID_ + n0, HID_, tid);
        if (k0 + 32 < XF_)
            __builtin_prefetch(Wo + (size_t)(k0 + 32) * HID_ + n0 + (tid >> 3) * 8, 0, 1);
    };

    v8f acc[8] = {};
    stage(0, 0);
    const int NK = XF_ / 32;
    for (int kk = 0; kk < NK; ++kk) {
        const int cur = kk & 1;
        wait_async_then_barrier();
        if (kk + 1 < NK) stage(cur ^ 1, (kk + 1) * 32);
        wmma_strip(lA[cur], lBt[cur], mt, nh2, half, l16, acc);
    }

    // epilogue: bias + exact gelu, fp32 out
    #pragma unroll
    for (int j = 0; j < 8; ++j) {
        const int n    = (nh2 * 8 + j) * 16 + l16;
        const float bv = bo[n0 + n];
        #pragma unroll
        for (int r = 0; r < 8; ++r) {
            const int mrow = mt * 16 + (half ? r + 8 : r);
            const float vv = acc[j][r] + bv;
            out[(size_t)(m0 + mrow) * HID_ + n0 + n] =
                0.5f * vv * (1.0f + erff(vv * 0.70710678118654752f));
        }
    }
}

// ---------------------------------------------------------------------------
extern "C" void kernel_launch(void* const* d_in, const int* in_sizes, int n_in,
                              void* d_out, int out_size, void* d_ws, size_t ws_size,
                              hipStream_t stream)
{
    const float* hs = (const float*)d_in[0];   // (B,T,HID)
    const float* Wm = (const float*)d_in[1];   // (HID, 4096)
    const float* bm = (const float*)d_in[2];   // (4096,)
    const float* Wo = (const float*)d_in[3];   // (512, HID)
    const float* bo = (const float*)d_in[4];   // (HID,)
    float* out = (float*)d_out;                // (B,T,HID) fp32

    // workspace: m_norm bf16 (128 MiB) then x bf16 (16 MiB)
    __bf16* mnorm = (__bf16*)d_ws;
    __bf16* xbuf  = (__bf16*)((char*)d_ws + (size_t)B_ * T_ * NMAT_ * sizeof(__bf16));

    k_gemm1_fro <<<dim3((B_ * T_) / 64, NMAT_ / 256), 256, 0, stream>>>(hs, Wm, bm, mnorm);
    k_scan      <<<dim3(32),                          256, 0, stream>>>(mnorm, xbuf);
    k_gemm2_gelu<<<dim3((B_ * T_) / 64, HID_ / 256),  256, 0, stream>>>(xbuf, Wo, bo, out);
}
